// MetaEmbedding_Classifier_5437428596737
// MI455X (gfx1250) — compile-verified, matrix operand order checked by reading, not run on hardware
//
#include <hip/hip_runtime.h>
#include <hip/hip_bf16.h>

typedef __bf16 bf16_t;
typedef __attribute__((ext_vector_type(16))) __bf16 v16bf;
typedef __attribute__((ext_vector_type(8)))  __bf16 v8bf;
typedef __attribute__((ext_vector_type(8)))  float  v8f;
typedef __attribute__((ext_vector_type(4)))  int    v4i;

#define AS1 __attribute__((address_space(1)))
#define AS3 __attribute__((address_space(3)))

#define BATCH 8192
#define FD    2048
#define NC    1000
#define NCP   1008   // 63*16, zero-padded class dim for WMMA tiles
#define CK    1024   // padded K for values_memory @ centroids

#if defined(__has_builtin)
#if __has_builtin(__builtin_amdgcn_global_load_async_to_lds_b128)
#define HAVE_ASYNC_LDS 1
#endif
#if __has_builtin(__builtin_amdgcn_s_wait_asynccnt)
#define HAVE_WAIT_ASYNC 1
#endif
#endif

// ---- bf16 WMMA fragment loader: lane l holds row (l&15); K chunks at
// (l>>4)*8 and 16+(l>>4)*8  (ISA 7.12.2 16-bit A/B layout). Two b128 loads.
__device__ __forceinline__ v16bf load_frag(const bf16_t* __restrict__ base,
                                           int row, int ld, int k0, int khalf) {
  const bf16_t* p = base + (size_t)row * ld + (k0 + khalf * 8);
  union { v16bf v; v8bf h[2]; } u;
  u.h[0] = *reinterpret_cast<const v8bf*>(p);
  u.h[1] = *reinterpret_cast<const v8bf*>(p + 16);
  return u.v;
}

__device__ __forceinline__ v8f wmma_bf16(v16bf a, v16bf b, v8f c) {
  return __builtin_amdgcn_wmma_f32_16x16x32_bf16(false, a, false, b,
                                                 (short)0, c, false, false);
}

// ---------------- prep kernels ----------------
__global__ void prep_x(const float* __restrict__ x, bf16_t* __restrict__ xbf,
                       float* __restrict__ xsq, float* __restrict__ out_direct) {
  int r = blockIdx.x, tid = threadIdx.x;
  __shared__ float sm[256];
  float s = 0.f;
  for (int c = tid; c < FD; c += 256) {
    size_t i = (size_t)r * FD + c;
    float v = x[i];
    xbf[i] = (bf16_t)v;
    out_direct[i] = v;          // direct_feature = x
    s += v * v;
  }
  sm[tid] = s; __syncthreads();
  for (int k = 128; k > 0; k >>= 1) { if (tid < k) sm[tid] += sm[tid + k]; __syncthreads(); }
  if (tid == 0) xsq[r] = sm[0];
}

__global__ void prep_rows(const float* __restrict__ cent, const float* __restrict__ Whall,
                          const float* __restrict__ bhall, const float* __restrict__ Wcos,
                          bf16_t* __restrict__ centbf, bf16_t* __restrict__ whallbf,
                          bf16_t* __restrict__ ewbf, float* __restrict__ csq,
                          float* __restrict__ bhallp) {
  int r = blockIdx.x, tid = threadIdx.x;     // r < 1008
  __shared__ float sm[256];
  const bool valid = r < NC;
  float sc = 0.f, sw = 0.f;
  for (int c = tid; c < FD; c += 256) {
    size_t i = (size_t)r * FD + c;
    float cv = valid ? cent[i]  : 0.f;
    float hv = valid ? Whall[i] : 0.f;
    float wv = valid ? Wcos[i]  : 0.f;
    centbf[i]  = (bf16_t)cv;
    whallbf[i] = (bf16_t)hv;
    sc += cv * cv; sw += wv * wv;
  }
  sm[tid] = sc; __syncthreads();
  for (int k = 128; k > 0; k >>= 1) { if (tid < k) sm[tid] += sm[tid + k]; __syncthreads(); }
  float csq_r = sm[0]; __syncthreads();
  sm[tid] = sw; __syncthreads();
  for (int k = 128; k > 0; k >>= 1) { if (tid < k) sm[tid] += sm[tid + k]; __syncthreads(); }
  float invn = valid ? rsqrtf(fmaxf(sm[0], 1e-30f)) : 0.f;
  for (int c = tid; c < FD; c += 256) {
    size_t i = (size_t)r * FD + c;
    float wv = valid ? Wcos[i] : 0.f;
    ewbf[i] = (bf16_t)(wv * invn);           // W_cos / ||row||
  }
  if (tid == 0) { csq[r] = csq_r; bhallp[r] = valid ? bhall[r] : 0.f; }
}

__global__ void prep_centT(const float* __restrict__ cent, bf16_t* __restrict__ centT) {
  int i = blockIdx.x * 256 + threadIdx.x;    // 2048*1024 elems
  int d = i >> 10, k = i & 1023;
  centT[i] = (k < NC) ? (bf16_t)cent[(size_t)k * FD + d] : (bf16_t)0.f;
}

__global__ void prep_wsel(const float* __restrict__ w, bf16_t* __restrict__ wb) {
  size_t i = (size_t)blockIdx.x * 256 + threadIdx.x;
  wb[i] = (bf16_t)w[i];
}

// ---------------- K1: dual GEMM (x@cent^T, x@W_hall^T) + min-dist + row-max ----------------
__global__ void __launch_bounds__(128)
k1_dual_gemm(const bf16_t* __restrict__ xbf, const bf16_t* __restrict__ centbf,
             const bf16_t* __restrict__ whallbf, const float* __restrict__ csq,
             const float* __restrict__ bhallp, const float* __restrict__ xsq,
             float* __restrict__ Hws, float* __restrict__ rowmax,
             float* __restrict__ reach) {
  __shared__ __align__(16) unsigned char smem[16 * FD * 2];   // 64 KB A-block, reused for reduce
  bf16_t* As = (bf16_t*)smem;
  const int tid = threadIdx.x;
  const int lane = tid & 31, widx = tid >> 5;
  const int r0 = blockIdx.x * 16;
  const int m = lane & 15, khalf = lane >> 4;

  // stage 16 x 2048 bf16 A rows into LDS; async memory->LDS path when available
  for (int idx = tid; idx < 16 * (FD / 8); idx += 128) {
    int row = idx / (FD / 8), ch = idx % (FD / 8);
    const v8bf* src = reinterpret_cast<const v8bf*>(xbf + (size_t)(r0 + row) * FD) + ch;
    v8bf* dst = reinterpret_cast<v8bf*>(As) + idx;
#ifdef HAVE_ASYNC_LDS
    __builtin_amdgcn_global_load_async_to_lds_b128(
        (AS1 v4i*)(v8bf*)src, (AS3 v4i*)dst, 0, 0);
#else
    *dst = *src;
#endif
  }
#ifdef HAVE_ASYNC_LDS
#ifdef HAVE_WAIT_ASYNC
  __builtin_amdgcn_s_wait_asynccnt(0);
#else
  asm volatile("s_wait_asynccnt 0x0" ::: "memory");
#endif
#endif
  __syncthreads();

  float xsq_r[8], dmin[8], hmax[8];
#pragma unroll
  for (int j = 0; j < 8; ++j) {
    xsq_r[j] = xsq[r0 + j + 8 * khalf];
    dmin[j] = 3.4e38f; hmax[j] = -3.4e38f;
  }

  for (int t = widx; t < NCP / 16; t += 4) {   // waves split the 63 N tiles
    const int nrow = t * 16 + m;
    v8f accS = {}, accH = {};
    for (int k0 = 0; k0 < FD; k0 += 32) {
      v16bf a  = load_frag(As, m, FD, k0, khalf);          // LDS
      v16bf bS = load_frag(centbf,  nrow, FD, k0, khalf);  // L2-resident
      v16bf bH = load_frag(whallbf, nrow, FD, k0, khalf);
      accS = wmma_bf16(a, bS, accS);
      accH = wmma_bf16(a, bH, accH);
    }
    const float csq_n = csq[nrow];
    const float bh_n  = bhallp[nrow];
    const bool nvalid = nrow < NC;
#pragma unroll
    for (int j = 0; j < 8; ++j) {
      const int M = r0 + j + 8 * khalf;
      float h = accH[j] + bh_n;
      Hws[(size_t)M * CK + nrow] = h;
      if (nvalid) {
        float d2 = xsq_r[j] + csq_n - 2.f * accS[j];
        dmin[j] = fminf(dmin[j], d2);
        hmax[j] = fmaxf(hmax[j], h);
      }
    }
  }
  // reduce across the 16-lane N groups
#pragma unroll
  for (int j = 0; j < 8; ++j)
    for (int msk = 1; msk < 16; msk <<= 1) {
      dmin[j] = fminf(dmin[j], __shfl_xor(dmin[j], msk, 32));
      hmax[j] = fmaxf(hmax[j], __shfl_xor(hmax[j], msk, 32));
    }
  __syncthreads();                       // A-block dead; reuse smem
  float* redMin = (float*)smem;
  float* redMax = redMin + 64;
  if (m == 0) {
#pragma unroll
    for (int j = 0; j < 8; ++j) {
      redMin[widx * 16 + khalf * 8 + j] = dmin[j];
      redMax[widx * 16 + khalf * 8 + j] = hmax[j];
    }
  }
  __syncthreads();
  if (tid < 16) {
    float dm = 3.4e38f, hm = -3.4e38f;
    for (int w = 0; w < 4; ++w) {
      dm = fminf(dm, redMin[w * 16 + tid]);
      hm = fmaxf(hm, redMax[w * 16 + tid]);
    }
    reach[r0 + tid]  = 10.0f / sqrtf(fmaxf(dm, 0.0f));
    rowmax[r0 + tid] = hm;
  }
}

// ---------------- K2: row softmax (unnormalized exp -> bf16; 1/sum folded into GEMM3) ---------
__global__ void softmax_rows(const float* __restrict__ Hws, const float* __restrict__ rowmax,
                             bf16_t* __restrict__ vmem, float* __restrict__ invsum) {
  int r = blockIdx.x, tid = threadIdx.x;
  __shared__ float sm[256];
  const float mx = rowmax[r];
  float s = 0.f;
  for (int c = tid; c < CK; c += 256) {
    size_t i = (size_t)r * CK + c;
    float e = 0.f;
    if (c < NC) e = __expf(Hws[i] - mx);
    vmem[i] = (bf16_t)e;
    s += e;
  }
  sm[tid] = s; __syncthreads();
  for (int k = 128; k > 0; k >>= 1) { if (tid < k) sm[tid] += sm[tid + k]; __syncthreads(); }
  if (tid == 0) invsum[r] = 1.f / sm[0];
}

// ---------------- K3: (V@cent) + (x@W_sel^T) fused -> infused, fused-norm ----------------
__global__ void __launch_bounds__(128)
k3_fuse(const bf16_t* __restrict__ vmem, const bf16_t* __restrict__ centT,
        const bf16_t* __restrict__ xbf, const bf16_t* __restrict__ wselbf,
        const float* __restrict__ bsel, const float* __restrict__ invsum,
        const float* __restrict__ reach, const float* __restrict__ x,
        float* __restrict__ out_inf, float* __restrict__ scale) {
  __shared__ float red[64];
  const int tid = threadIdx.x;
  const int lane = tid & 31, widx = tid >> 5;
  const int r0 = blockIdx.x * 16;
  const int m = lane & 15, khalf = lane >> 4;

  float iv[8], rch[8], ns[8];
#pragma unroll
  for (int j = 0; j < 8; ++j) {
    int M = r0 + j + 8 * khalf;
    iv[j] = invsum[M]; rch[j] = reach[M]; ns[j] = 0.f;
  }

  for (int t = widx; t < FD / 16; t += 4) {
    const int nrow = t * 16 + m;
    v8f aM = {}, aS = {};
    for (int k0 = 0; k0 < CK; k0 += 32) {                       // values_memory @ centroids
      v16bf a = load_frag(vmem,  r0 + m, CK, k0, khalf);
      v16bf b = load_frag(centT, nrow,   CK, k0, khalf);
      aM = wmma_bf16(a, b, aM);
    }
    for (int k0 = 0; k0 < FD; k0 += 32) {                       // x @ W_sel^T
      v16bf a = load_frag(xbf,    r0 + m, FD, k0, khalf);
      v16bf b = load_frag(wselbf, nrow,   FD, k0, khalf);
      aS = wmma_bf16(a, b, aS);
    }
    const float bs = bsel[nrow];
#pragma unroll
    for (int j = 0; j < 8; ++j) {
      const int M = r0 + j + 8 * khalf;
      float mem = aM[j] * iv[j];
      float sel = tanhf(aS[j] + bs);
      float inf = sel * mem;
      size_t oi = (size_t)M * FD + nrow;
      out_inf[oi] = inf;                                        // infused_feature
      float f = rch[j] * (x[oi] + inf);                         // fused (recomputed in K4)
      ns[j] += f * f;
    }
  }
#pragma unroll
  for (int j = 0; j < 8; ++j)
    for (int msk = 1; msk < 16; msk <<= 1)
      ns[j] += __shfl_xor(ns[j], msk, 32);
  if (m == 0) {
#pragma unroll
    for (int j = 0; j < 8; ++j) red[widx * 16 + khalf * 8 + j] = ns[j];
  }
  __syncthreads();
  if (tid < 16) {
    float s = 0.f;
    for (int w = 0; w < 4; ++w) s += red[w * 16 + tid];
    scale[r0 + tid] = 1.f / (1.f + sqrtf(s));                  // ex = fused/(1+||fused||)
  }
}

// ---------------- K4: ex (bf16) = scale * reach * (x + infused) ----------------
__global__ void k4_ex(const float* __restrict__ x, const float* __restrict__ out_inf,
                      const float* __restrict__ reach, const float* __restrict__ scale,
                      bf16_t* __restrict__ exbf) {
  size_t i = (size_t)blockIdx.x * 256 + threadIdx.x;
  int r = (int)(i >> 11);
  exbf[i] = (bf16_t)(scale[r] * reach[r] * (x[i] + out_inf[i]));
}

// ---------------- K5: logits = 16 * ex @ ew^T ----------------
__global__ void __launch_bounds__(256)
k5_logits(const bf16_t* __restrict__ exbf, const bf16_t* __restrict__ ewbf,
          float* __restrict__ out_logits) {
  const int tid = threadIdx.x;
  const int lane = tid & 31, widx = tid >> 5;
  const int r0 = (blockIdx.x * 8 + widx) * 16;
  const int m = lane & 15, khalf = lane >> 4;
  for (int t = 0; t < NCP / 16; ++t) {
    const int nrow = t * 16 + m;
    if (t + 1 < NCP / 16)
      __builtin_prefetch(ewbf + (size_t)(nrow + 16) * FD, 0, 1);  // global_prefetch
    v8f acc = {};
    for (int k0 = 0; k0 < FD; k0 += 32) {
      v16bf a = load_frag(exbf, r0 + m, FD, k0, khalf);
      v16bf b = load_frag(ewbf, nrow,   FD, k0, khalf);
      acc = wmma_bf16(a, b, acc);
    }
    if (nrow < NC) {
#pragma unroll
      for (int j = 0; j < 8; ++j)
        out_logits[(size_t)(r0 + j + 8 * khalf) * NC + nrow] = 16.f * acc[j];
    }
  }
}

// ---------------- host ----------------
extern "C" void kernel_launch(void* const* d_in, const int* in_sizes, int n_in,
                              void* d_out, int out_size, void* d_ws, size_t ws_size,
                              hipStream_t stream) {
  (void)in_sizes; (void)n_in; (void)out_size; (void)ws_size;
  const float* x     = (const float*)d_in[0];
  const float* cent  = (const float*)d_in[1];
  const float* Whall = (const float*)d_in[2];
  const float* bhall = (const float*)d_in[3];
  const float* Wsel  = (const float*)d_in[4];
  const float* bsel  = (const float*)d_in[5];
  const float* Wcos  = (const float*)d_in[6];

  float* out_logits = (float*)d_out;
  float* out_direct = out_logits + (size_t)BATCH * NC;
  float* out_inf    = out_direct + (size_t)BATCH * FD;

  unsigned char* ws = (unsigned char*)d_ws;
  size_t off = 0;
  auto alloc = [&](size_t bytes) -> void* {
    void* p = ws + off; off += (bytes + 255) & ~(size_t)255; return p;
  };
  bf16_t* xbf     = (bf16_t*)alloc((size_t)BATCH * FD * 2);   // 32 MB
  bf16_t* centbf  = (bf16_t*)alloc((size_t)NCP * FD * 2);     //  4 MB (zero-padded rows)
  bf16_t* whallbf = (bf16_t*)alloc((size_t)NCP * FD * 2);     //  4 MB
  bf16_t* ewbf    = (bf16_t*)alloc((size_t)NCP * FD * 2);     //  4 MB (normalized W_cos)
  bf16_t* centT   = (bf16_t*)alloc((size_t)FD * CK * 2);      //  4 MB (transposed, padded K)
  bf16_t* wselbf  = (bf16_t*)alloc((size_t)FD * FD * 2);      //  8 MB
  bf16_t* vmem    = (bf16_t*)alloc((size_t)BATCH * CK * 2);   // 16 MB
  float*  Hws     = (float*)alloc((size_t)BATCH * CK * 4);    // 32 MB, reused as exbf
  float*  csq     = (float*)alloc(NCP * 4);
  float*  bhallp  = (float*)alloc(NCP * 4);
  float*  xsq     = (float*)alloc(BATCH * 4);
  float*  rowmax  = (float*)alloc(BATCH * 4);
  float*  reach   = (float*)alloc(BATCH * 4);
  float*  invsum  = (float*)alloc(BATCH * 4);
  float*  scale   = (float*)alloc(BATCH * 4);
  bf16_t* exbf    = (bf16_t*)Hws;   // Hws dead after K2; same 33.5 MB footprint

  prep_x    <<<BATCH, 256, 0, stream>>>(x, xbf, xsq, out_direct);
  prep_rows <<<NCP,   256, 0, stream>>>(cent, Whall, bhall, Wcos, centbf, whallbf, ewbf, csq, bhallp);
  prep_centT<<<(FD * CK) / 256, 256, 0, stream>>>(cent, centT);
  prep_wsel <<<(FD * FD) / 256, 256, 0, stream>>>(Wsel, wselbf);

  k1_dual_gemm<<<BATCH / 16, 128, 0, stream>>>(xbf, centbf, whallbf, csq, bhallp, xsq,
                                               Hws, rowmax, reach);
  softmax_rows<<<BATCH, 256, 0, stream>>>(Hws, rowmax, vmem, invsum);
  k3_fuse     <<<BATCH / 16, 128, 0, stream>>>(vmem, centT, xbf, wselbf, bsel, invsum,
                                               reach, x, out_inf, scale);
  k4_ex       <<<(BATCH * FD) / 256, 256, 0, stream>>>(x, out_inf, reach, scale, exbf);
  k5_logits   <<<BATCH / 16 / 8, 256, 0, stream>>>(exbf, ewbf, out_logits);
}
